// KeyNetDetector_80917183857272
// MI455X (gfx1250) — matrix operand order repaired and checked by reading, not compiled, not run on hardware
//
#include <hip/hip_runtime.h>
#include <math.h>

typedef _Float16 f16;
typedef __attribute__((ext_vector_type(16))) _Float16 v16h;
typedef __attribute__((ext_vector_type(8)))  _Float16 v8h;
typedef __attribute__((ext_vector_type(8)))  float    v8f;
typedef __attribute__((ext_vector_type(4)))  unsigned u32x4;
typedef __attribute__((ext_vector_type(8)))  int      i32x8;
typedef __attribute__((ext_vector_type(4)))  int      i32x4;

#define DEV __device__ __forceinline__

DEV int iclamp(int v, int lo, int hi) { return v < lo ? lo : (v > hi ? hi : v); }
DEV int ireflect(int i, int n) { if (i < 0) i = -i; if (i >= n) i = 2 * n - 2 - i; return iclamp(i, 0, n - 1); }

// ---------------------------------------------------------------- utilities
__global__ void k_zero_int(int* p) { *p = 0; }

__global__ void k_f32tof16(const float* __restrict__ in, f16* __restrict__ out, int n) {
  int t = blockIdx.x * 256 + threadIdx.x;
  if (t < n) out[t] = (f16)in[t];
}

// bilinear resize (jax.image.resize 'bilinear', half-pixel centers), C channels, f16->f16
__global__ void k_resize(const f16* __restrict__ in, int Hs, int Ws,
                         f16* __restrict__ out, int Hd, int Wd, int C) {
  int t = blockIdx.x * 256 + threadIdx.x;
  int tot = C * Hd * Wd;
  if (t >= tot) return;
  int c = t / (Hd * Wd);
  int r = t - c * Hd * Wd;
  int y = r / Wd, x = r - y * Wd;
  float sy = (float)Hs / (float)Hd, sx = (float)Ws / (float)Wd;
  float fy = ((float)y + 0.5f) * sy - 0.5f;
  float fx = ((float)x + 0.5f) * sx - 0.5f;
  fy = fminf(fmaxf(fy, 0.f), (float)(Hs - 1));
  fx = fminf(fmaxf(fx, 0.f), (float)(Ws - 1));
  int y0 = (int)fy, x0 = (int)fx;
  int y1 = (y0 + 1 < Hs) ? y0 + 1 : Hs - 1;
  int x1 = (x0 + 1 < Ws) ? x0 + 1 : Ws - 1;
  float wy = fy - (float)y0, wx = fx - (float)x0;
  const f16* src = in + (size_t)c * Hs * Ws;
  float v00 = (float)src[(size_t)y0 * Ws + x0], v01 = (float)src[(size_t)y0 * Ws + x1];
  float v10 = (float)src[(size_t)y1 * Ws + x0], v11 = (float)src[(size_t)y1 * Ws + x1];
  float v = v00 * (1.f - wy) * (1.f - wx) + v01 * (1.f - wy) * wx +
            v10 * wy * (1.f - wx) + v11 * wy * wx;
  out[t] = (f16)v;
}

// 5x5 gaussian (sigma=1), reflect padding
__global__ void k_gauss5(const f16* __restrict__ in, int H, int W, f16* __restrict__ out) {
  int t = blockIdx.x * 256 + threadIdx.x;
  if (t >= H * W) return;
  int y = t / W, x = t - y * W;
  const float g[5] = {0.054488685f, 0.24420134f, 0.40261996f, 0.24420134f, 0.054488685f};
  float acc = 0.f;
#pragma unroll
  for (int i = 0; i < 5; ++i) {
    int yy = ireflect(y + i - 2, H);
#pragma unroll
    for (int j = 0; j < 5; ++j) {
      int xx = ireflect(x + j - 2, W);
      acc += g[i] * g[j] * (float)in[(size_t)yy * W + xx];
    }
  }
  out[t] = (f16)acc;
}

// ------------------------------------------------- handcrafted 10ch features
__global__ void k_handcrafted(const f16* __restrict__ img, int H, int W, f16* __restrict__ out) {
  int t = blockIdx.x * 256 + threadIdx.x;
  if (t >= H * W) return;
  int y = t / W, x = t - y * W;
  float p[5][5];
#pragma unroll
  for (int i = 0; i < 5; ++i) {
    int yy = iclamp(y + i - 2, 0, H - 1);
#pragma unroll
    for (int j = 0; j < 5; ++j) {
      int xx = iclamp(x + j - 2, 0, W - 1);
      p[i][j] = (float)img[(size_t)yy * W + xx];
    }
  }
  float dx3[3][3], dy3[3][3];
#pragma unroll
  for (int a = 0; a < 3; ++a)
#pragma unroll
    for (int b = 0; b < 3; ++b) {
      int cy = 1 + a, cx = 1 + b;
      dx3[a][b] = (p[cy - 1][cx + 1] + 2.f * p[cy][cx + 1] + p[cy + 1][cx + 1]
                 - p[cy - 1][cx - 1] - 2.f * p[cy][cx - 1] - p[cy + 1][cx - 1]) * 0.125f;
      dy3[a][b] = (p[cy + 1][cx - 1] + 2.f * p[cy + 1][cx] + p[cy + 1][cx + 1]
                 - p[cy - 1][cx - 1] - 2.f * p[cy - 1][cx] - p[cy - 1][cx + 1]) * 0.125f;
    }
  float dx = dx3[1][1], dy = dy3[1][1];
  float dxx = (dx3[0][2] + 2.f * dx3[1][2] + dx3[2][2] - dx3[0][0] - 2.f * dx3[1][0] - dx3[2][0]) * 0.125f;
  float dxy = (dx3[2][0] + 2.f * dx3[2][1] + dx3[2][2] - dx3[0][0] - 2.f * dx3[0][1] - dx3[0][2]) * 0.125f;
  float dyy = (dy3[2][0] + 2.f * dy3[2][1] + dy3[2][2] - dy3[0][0] - 2.f * dy3[0][1] - dy3[0][2]) * 0.125f;
  float ch[10] = {dx, dy, dx * dx, dy * dy, dx * dy, dxy, dxy * dxy, dxx, dyy, dxx * dyy};
  size_t HW = (size_t)H * W;
#pragma unroll
  for (int c = 0; c < 10; ++c) out[c * HW + t] = (f16)ch[c];
}

// --------------------------------------------- weight packing (BN folded)
// K-order: k = tap*CPAD + ci  (tap = ky*5+kx in 0..24, channels fastest, padded to CPAD)
// B-fragment (32x16 f16): lanes 0-15: N=lane, K=h<8?h:h+8; lanes 16-31: N=lane-16, K=h<8?h+8:h+16
__global__ __launch_bounds__(512)
void k_pack(const float* w0, const float* b0, const float* g0, const float* be0, const float* m0, const float* v0,
            const float* w1, const float* b1, const float* g1, const float* be1, const float* m1, const float* v1,
            const float* w2, const float* b2, const float* g2, const float* be2, const float* m2, const float* v2,
            const float* wl, const float* bl,
            f16* __restrict__ wpack, float* __restrict__ bias) {
  const int choff[4] = {0, 13, 26, 39};
  int blk = blockIdx.x;  // 0..63
  int layer, chunk;
  if (blk < 13)      { layer = 0; chunk = blk; }
  else if (blk < 26) { layer = 1; chunk = blk - 13; }
  else if (blk < 39) { layer = 2; chunk = blk - 26; }
  else               { layer = 3; chunk = blk - 39; }
  const float *W_, *B_, *G_, *BE_, *M_, *V_;
  int Cin, Cout, CP;
  if (layer == 0)      { W_ = w0; B_ = b0; G_ = g0; BE_ = be0; M_ = m0; V_ = v0; Cin = 10; Cout = 8; CP = 16; }
  else if (layer == 1) { W_ = w1; B_ = b1; G_ = g1; BE_ = be1; M_ = m1; V_ = v1; Cin = 8;  Cout = 8; CP = 16; }
  else if (layer == 2) { W_ = w2; B_ = b2; G_ = g2; BE_ = be2; M_ = m2; V_ = v2; Cin = 8;  Cout = 8; CP = 16; }
  else                 { W_ = wl; B_ = bl; G_ = 0;  BE_ = 0;   M_ = 0;  V_ = 0;  Cin = 24; Cout = 1; CP = 32; }

  int lane = threadIdx.x >> 4;
  int h    = threadIdx.x & 15;
  int n  = lane & 15;
  int kl = (h & 7) + ((h >= 8) ? 16 : 0) + ((lane >= 16) ? 8 : 0);
  int k  = chunk * 32 + kl;
  int tap = k / CP, ci = k & (CP - 1);
  float val = 0.f;
  if (n < Cout && tap < 25 && ci < Cin) {
    int ky = tap / 5, kx = tap % 5;
    float wv = W_[(((size_t)n * Cin + ci) * 5 + ky) * 5 + kx];
    float s = 1.f;
    if (layer < 3) s = G_[n] * rsqrtf(V_[n] + 1e-5f);
    val = wv * s;
  }
  wpack[((size_t)(choff[layer] + chunk) * 32 + lane) * 16 + h] = (f16)val;

  if (chunk == 0 && threadIdx.x < 16) {
    int nn = threadIdx.x;
    float bb = 0.f;
    if (nn < Cout) {
      if (layer < 3) {
        float s = G_[nn] * rsqrtf(V_[nn] + 1e-5f);
        bb = (B_[nn] - M_[nn]) * s + BE_[nn];
      } else bb = B_[0];
    }
    bias[layer * 16 + nn] = bb;
  }
}

// -------------------------------- 5x5 conv as implicit GEMM on WMMA units
// Tile 16x8 outputs per WG (8 waves, one row/wave). Input staged NHWC-CPAD in
// LDS so each lane's A fragment = two contiguous 16B ds_load_b128. Packed B
// fragments are broadcast into LDS by a single TDM tensor_load (wave 0).
template <int CPAD>
__global__ __launch_bounds__(256)
void k_conv5x5_wmma(const f16* __restrict__ in, int Cin, int H, int W,
                    const f16* __restrict__ wpack, const float* __restrict__ bias,
                    int nchunks, int Cout,
                    f16* __restrict__ out_h, float* __restrict__ out_f, int relu) {
  const int tx   = blockIdx.x * 16;
  const int ty   = blockIdx.y * 8;
  const int lane = threadIdx.x & 31;
  const int wave = threadIdx.x >> 5;

  __shared__ __align__(16) f16 a_smem[13 * 20 * 32];  // 16.6 KB, NHWC-CPAD + halo
  __shared__ __align__(16) f16 b_smem[25 * 512];      // 25.6 KB packed B fragments

  // ---- TDM broadcast of this layer's B fragments into LDS (wave 0 only) ----
  if (threadIdx.x < 32) {
    unsigned lds_b = (unsigned)(size_t)(void*)b_smem;
    unsigned long long ga = (unsigned long long)(size_t)wpack;
    unsigned nelem = (unsigned)nchunks * 512u;  // 2-byte elements, <= 12800
    u32x4 g0;
    g0[0] = 1u;                                          // count=1
    g0[1] = lds_b;                                       // lds_addr
    g0[2] = (unsigned)(ga & 0xffffffffu);                // global_addr lo
    g0[3] = (unsigned)((ga >> 32) & 0x1ffffffu) | (2u << 30);  // addr hi | type=2
    i32x8 g1;
    g1[0] = (int)(1u << 16);                             // data_size=1 (2B)
    g1[1] = (int)((nelem & 0xffffu) << 16);              // tensor_dim0[15:0]
    g1[2] = (int)((nelem >> 16) | (1u << 16));           // tensor_dim0[31:16] | tensor_dim1=1
    g1[3] = (int)((nelem & 0xffffu) << 16);              // tile_dim0
    g1[4] = 1;                                           // tile_dim1=1, tile_dim2=0
    g1[5] = (int)nelem;                                  // tensor_dim0_stride lo
    g1[6] = 0;
    g1[7] = 0;
    i32x4 z4 = {0, 0, 0, 0};
    i32x8 z8 = {0, 0, 0, 0, 0, 0, 0, 0};
    __builtin_amdgcn_tensor_load_to_lds(g0, g1, z4, z4, z8, 0);
  }

  __builtin_prefetch(in, 0, 1);

  // zero-fill A stage (padded channels must be 0.0, never garbage/NaN)
  {
    v8h z = {};
    for (int t = threadIdx.x; t < (13 * 20 * 32) / 8; t += 256)
      ((v8h*)a_smem)[t] = z;
  }
  __syncthreads();
  // stage input tile (halo 2, 13 rows x 20 cols) into NHWC-CPAD layout
  const int tot = Cin * 260;
  for (int t = threadIdx.x; t < tot; t += 256) {
    int c = t / 260;
    int r = t - c * 260;
    int rr = r / 20, cc = r - rr * 20;
    int gy = ty + rr - 2, gx = tx + cc - 2;
    f16 v = (f16)0.f;
    if (gy >= 0 && gy < H && gx >= 0 && gx < W)
      v = in[(size_t)c * H * W + (size_t)gy * W + gx];
    a_smem[(rr * 20 + cc) * CPAD + c] = v;
  }
  if (threadIdx.x < 32) __builtin_amdgcn_s_wait_tensorcnt(0);
  __syncthreads();

  const int m  = lane & 15;
  const int kb = (lane >= 16) ? 8 : 0;
  v8f acc = {};
  for (int ch = 0; ch < nchunks; ++ch) {
    const f16 *pa, *pb;
    if (CPAD == 16) {
      int t0 = 2 * ch, t1 = 2 * ch + 1;       // two taps per 32-K chunk
      int ky0 = t0 / 5, kx0 = t0 - 5 * ky0;
      int ky1 = t1 / 5, kx1 = t1 - 5 * ky1;
      pa = &a_smem[((wave + ky0) * 20 + m + kx0) * 16 + kb];
      pb = &a_smem[((wave + ky1) * 20 + m + kx1) * 16 + kb];
    } else {
      int ky = ch / 5, kx = ch - 5 * ky;      // one tap per chunk (CPAD=32)
      const f16* base = &a_smem[((wave + ky) * 20 + m + kx) * 32];
      pa = base + kb;
      pb = base + 16 + kb;
    }
    v8h lo = *(const v8h*)pa;   // ds_load_b128
    v8h hi = *(const v8h*)pb;   // ds_load_b128
    v16h a;
#pragma unroll
    for (int i = 0; i < 8; ++i) { a[i] = lo[i]; a[8 + i] = hi[i]; }
    v16h b = *(const v16h*)(b_smem + ((size_t)ch * 32 + lane) * 16);
    acc = __builtin_amdgcn_wmma_f32_16x16x32_f16(false, a, false, b, (short)0, acc, false, false);
  }

  int n = lane & 15;
  int mbase = (lane < 16) ? 0 : 8;
  if (n < Cout) {
    float bv = bias[n];
    int py = ty + wave;
#pragma unroll
    for (int r = 0; r < 8; ++r) {
      float vv = acc[r] + bv;
      if (relu) vv = fmaxf(vv, 0.f);
      int px = tx + r + mbase;
      if (px < W && py < H) {
        size_t o = (size_t)n * H * W + (size_t)py * W + px;
        if (out_f) out_f[o] = vv;
        else       out_h[o] = (f16)vv;
      }
    }
  }
}

// -------------------------------- border removal + 15x15 NMS + compaction
__global__ void k_nms_collect(const float* __restrict__ sc, int H, int W,
                              float* __restrict__ cs, int* __restrict__ ci,
                              int* __restrict__ cnt, int cap) {
  int t = blockIdx.x * 256 + threadIdx.x;
  if (t >= H * W) return;
  int x = t % W, y = t / W;
  if (x < 15 || y < 15 || x >= W - 15 || y >= H - 15) return;
  float v = sc[t];
  if (v <= 0.f) return;
  float m = v;
  for (int dy = -7; dy <= 7; ++dy) {
    int yy = y + dy;
    if (yy < 0 || yy >= H) continue;
    for (int dx = -7; dx <= 7; ++dx) {
      int xx = x + dx;
      if (xx < 0 || xx >= W) continue;
      float u = (xx < 15 || yy < 15 || xx >= W - 15 || yy >= H - 15) ? 0.f : sc[(size_t)yy * W + xx];
      m = fmaxf(m, u);
    }
  }
  if (v >= m) {
    int p = atomicAdd(cnt, 1);
    if (p < cap) { cs[p] = v; ci[p] = t; }
  }
}

// ------------------------------------ per-level top-k + LAF construction
__global__ __launch_bounds__(1024)
void k_topk_level(const float* __restrict__ cs, const int* __restrict__ ci,
                  const int* __restrict__ cnt_p, int cap, int nkp, int w,
                  float fx, float fy, float* __restrict__ resp, float* __restrict__ lafs) {
  __shared__ float s[4096];
  __shared__ int   si[4096];
  __shared__ float rmax[1024];
  __shared__ int   rpos[1024];
  int n = *cnt_p; if (n > cap) n = cap;
  for (int t = threadIdx.x; t < 4096; t += 1024) {
    s[t]  = (t < n) ? cs[t] : -1.f;
    si[t] = (t < n) ? ci[t] : 0;
  }
  __syncthreads();
  for (int it = 0; it < nkp; ++it) {
    float bm = -2.f; int bp = 0;
    for (int t = threadIdx.x; t < 4096; t += 1024)
      if (s[t] > bm) { bm = s[t]; bp = t; }
    rmax[threadIdx.x] = bm; rpos[threadIdx.x] = bp;
    __syncthreads();
    for (int off = 512; off > 0; off >>= 1) {
      if (threadIdx.x < off && rmax[threadIdx.x + off] > rmax[threadIdx.x]) {
        rmax[threadIdx.x] = rmax[threadIdx.x + off];
        rpos[threadIdx.x] = rpos[threadIdx.x + off];
      }
      __syncthreads();
    }
    if (threadIdx.x == 0) {
      int j = rpos[0];
      float v = rmax[0]; if (v < 0.f) v = 0.f;
      int idx = si[j];
      float xx = (float)(idx % w) * fx;
      float yy = (float)(idx / w) * fy;
      float scv = 0.5f * (fx + fy) * 22.0f;
      resp[it] = v;
      float* L = lafs + (size_t)it * 6;
      L[0] = scv; L[1] = 0.f; L[2] = xx; L[3] = 0.f; L[4] = scv; L[5] = yy;
      s[j] = -1.f;
    }
    __syncthreads();
  }
}

// ------------------------------------------------- global final top-2048
#define MAXT 12288
__global__ __launch_bounds__(1024)
void k_topk_final(const float* __restrict__ resp, const float* __restrict__ lafs,
                  int T, float* __restrict__ out) {
  __shared__ float s[MAXT];
  __shared__ float rmax[1024];
  __shared__ int   rpos[1024];
  for (int t = threadIdx.x; t < MAXT; t += 1024) s[t] = (t < T) ? resp[t] : -1e30f;
  __syncthreads();
  for (int it = 0; it < 2048; ++it) {
    float bm = -1e31f; int bp = 0;
    for (int t = threadIdx.x; t < MAXT; t += 1024)
      if (s[t] > bm) { bm = s[t]; bp = t; }
    rmax[threadIdx.x] = bm; rpos[threadIdx.x] = bp;
    __syncthreads();
    for (int off = 512; off > 0; off >>= 1) {
      if (threadIdx.x < off && rmax[threadIdx.x + off] > rmax[threadIdx.x]) {
        rmax[threadIdx.x] = rmax[threadIdx.x + off];
        rpos[threadIdx.x] = rpos[threadIdx.x + off];
      }
      __syncthreads();
    }
    if (threadIdx.x == 0) {
      int j = rpos[0];
      const float* L = lafs + (size_t)j * 6;
      float* O = out + (size_t)it * 6;
      O[0] = L[0]; O[1] = L[1]; O[2] = L[2]; O[3] = L[3]; O[4] = L[4]; O[5] = L[5];
      out[2048 * 6 + it] = rmax[0];
      s[j] = -1e30f;
    }
    __syncthreads();
  }
}

// ============================================================ host driver
extern "C" void kernel_launch(void* const* d_in, const int* in_sizes, int n_in,
                              void* d_out, int out_size, void* d_ws, size_t ws_size,
                              hipStream_t stream) {
  (void)in_sizes; (void)n_in; (void)out_size; (void)ws_size;
  const float* img = (const float*)d_in[0];
  const float* w0 = (const float*)d_in[1];  const float* b0 = (const float*)d_in[2];
  const float* g0 = (const float*)d_in[3];  const float* be0 = (const float*)d_in[4];
  const float* m0 = (const float*)d_in[5];  const float* v0 = (const float*)d_in[6];
  const float* w1 = (const float*)d_in[7];  const float* b1 = (const float*)d_in[8];
  const float* g1 = (const float*)d_in[9];  const float* be1 = (const float*)d_in[10];
  const float* m1 = (const float*)d_in[11]; const float* v1 = (const float*)d_in[12];
  const float* w2 = (const float*)d_in[13]; const float* b2 = (const float*)d_in[14];
  const float* g2 = (const float*)d_in[15]; const float* be2 = (const float*)d_in[16];
  const float* m2 = (const float*)d_in[17]; const float* v2 = (const float*)d_in[18];
  const float* wl = (const float*)d_in[19]; const float* bl = (const float*)d_in[20];
  float* out = (float*)d_out;

  const int H = 512, W = 512;
  const size_t P = 724 * 724;

  size_t off = 0;
  auto alloc = [&](size_t bytes) -> size_t {
    size_t o = (off + 255) & ~(size_t)255;
    off = o + bytes;
    return o;
  };
  char* ws = (char*)d_ws;
  f16* img16  = (f16*)(ws + alloc(262144 * sizeof(f16)));
  f16* b_up   = (f16*)(ws + alloc(P * sizeof(f16)));
  f16* b_cur  = (f16*)(ws + alloc(P * sizeof(f16)));
  f16* b_nxt  = (f16*)(ws + alloc(P * sizeof(f16)));
  f16* b_tmp  = (f16*)(ws + alloc(P * sizeof(f16)));
  f16* b_s1   = (f16*)(ws + alloc(P * sizeof(f16)));
  f16* b_s2   = (f16*)(ws + alloc(P * sizeof(f16)));
  f16* b_hand = (f16*)(ws + alloc(10 * P * sizeof(f16)));
  f16* b_cA   = (f16*)(ws + alloc(8 * P * sizeof(f16)));
  f16* b_cB   = (f16*)(ws + alloc(8 * P * sizeof(f16)));
  f16* b_fe   = (f16*)(ws + alloc(8 * P * sizeof(f16)));
  f16* b_feat = (f16*)(ws + alloc(24 * P * sizeof(f16)));
  float* b_score = (float*)(ws + alloc(P * sizeof(float)));
  float* cand_s  = (float*)(ws + alloc(4096 * sizeof(float)));
  int*   cand_i  = (int*)(ws + alloc(4096 * sizeof(int)));
  int*   cand_c  = (int*)(ws + alloc(sizeof(int)));
  float* resp_all = (float*)(ws + alloc(MAXT * sizeof(float)));
  float* lafs_all = (float*)(ws + alloc(MAXT * 6 * sizeof(float)));
  f16*   wpack    = (f16*)(ws + alloc(64 * 512 * sizeof(f16)));
  float* biasbuf  = (float*)(ws + alloc(64 * sizeof(float)));

  const int choff[4]     = {0, 13, 26, 39};
  const int nchunks_l[4] = {13, 13, 13, 25};

  k_pack<<<64, 512, 0, stream>>>(w0, b0, g0, be0, m0, v0, w1, b1, g1, be1, m1, v1,
                                 w2, b2, g2, be2, m2, v2, wl, bl, wpack, biasbuf);
  k_f32tof16<<<(H * W + 255) / 256, 256, 0, stream>>>(img, img16, H * W);

  auto conv16 = [&](const f16* in, int Cin, int h, int w, int layer, int Cout, f16* oh) {
    dim3 g((w + 15) / 16, (h + 7) / 8);
    k_conv5x5_wmma<16><<<g, 256, 0, stream>>>(in, Cin, h, w, wpack + (size_t)choff[layer] * 512,
                                              biasbuf + layer * 16, nchunks_l[layer], Cout,
                                              oh, nullptr, 1);
  };
  auto conv_final = [&](const f16* in, int h, int w, float* of) {
    dim3 g((w + 15) / 16, (h + 7) / 8);
    k_conv5x5_wmma<32><<<g, 256, 0, stream>>>(in, 24, h, w, wpack + (size_t)choff[3] * 512,
                                              biasbuf + 3 * 16, nchunks_l[3], 1,
                                              nullptr, of, 1);
  };
  auto run_fe = [&](const f16* x, int h, int w, f16* outp) {
    k_handcrafted<<<(h * w + 255) / 256, 256, 0, stream>>>(x, h, w, b_hand);
    conv16(b_hand, 10, h, w, 0, 8, b_cA);
    conv16(b_cA,    8, h, w, 1, 8, b_cB);
    conv16(b_cB,    8, h, w, 2, 8, outp);
  };
  auto pyrdown = [&](const f16* src, int h, int w, f16* dst, int nh, int nw) {
    k_gauss5<<<(h * w + 255) / 256, 256, 0, stream>>>(src, h, w, b_tmp);
    k_resize<<<(nh * nw + 255) / 256, 256, 0, stream>>>(b_tmp, h, w, dst, nh, nw, 1);
  };
  auto run_keynet = [&](const f16* x, int h, int w) {
    size_t hw = (size_t)h * w;
    run_fe(x, h, w, b_feat);  // channels 0..7 directly
    int h1 = (int)(h / 1.2), w1 = (int)(w / 1.2);
    pyrdown(x, h, w, b_s1, h1, w1);
    run_fe(b_s1, h1, w1, b_fe);
    k_resize<<<(8 * h * w + 255) / 256, 256, 0, stream>>>(b_fe, h1, w1, b_feat + 8 * hw, h, w, 8);
    int h2 = (int)(h1 / 1.2), w2 = (int)(w1 / 1.2);
    pyrdown(b_s1, h1, w1, b_s2, h2, w2);
    run_fe(b_s2, h2, w2, b_fe);
    k_resize<<<(8 * h * w + 255) / 256, 256, 0, stream>>>(b_fe, h2, w2, b_feat + 16 * hw, h, w, 8);
    conv_final(b_feat, h, w, b_score);
  };
  auto detect = [&](const f16* x, int h, int w, float fx, float fy, int nkp, int roff) {
    run_keynet(x, h, w);
    k_zero_int<<<1, 1, 0, stream>>>(cand_c);
    k_nms_collect<<<(h * w + 255) / 256, 256, 0, stream>>>(b_score, h, w, cand_s, cand_i, cand_c, 4096);
    k_topk_level<<<1, 1024, 0, stream>>>(cand_s, cand_i, cand_c, 4096, nkp, w, fx, fy,
                                         resp_all + roff, lafs_all + (size_t)roff * 6);
  };

  // number of features per level (mirror reference)
  double fp = 2.0, tsum = 0.0;
  int nfpl[6];
  {
    double nfs[6];
    for (int i = 0; i < 6; ++i) {
      tsum += pow(fp, -(double)(i - 1));
      nfs[i] = 2048.0 * pow(fp, -(double)(i - 1));
    }
    for (int i = 0; i < 6; ++i) nfpl[i] = (int)(nfs[i] / tsum);
  }

  int roff = 0;
  {
    double uf = sqrt(2.0);
    int nh = (int)(H * uf), nw = (int)(W * uf);
    k_resize<<<(nh * nw + 255) / 256, 256, 0, stream>>>(img16, H, W, b_up, nh, nw, 1);
    int nkp = nfpl[0];
    detect(b_up, nh, nw, (float)W / nw, (float)H / nh, nkp, roff);
    roff += nkp;
  }
  {
    const f16* cur = img16;
    int ch = H, cw = W;
    for (int i = 0; i <= 4; ++i) {
      if (i > 0) {
        int nh = (int)(ch / sqrt(2.0)), nw = (int)(cw / sqrt(2.0));
        f16* dst = (i == 1) ? b_cur : b_nxt;
        pyrdown(cur, ch, cw, dst, nh, nw);
        cur = dst;
        ch = nh; cw = nw;
        if (i >= 2) { f16* t = b_cur; b_cur = b_nxt; b_nxt = t; }
      }
      float fx = (i == 0) ? 1.f : (float)W / cw;
      float fy = (i == 0) ? 1.f : (float)H / ch;
      int nkp = 0;
      for (int j = 0; j <= i + 1; ++j) nkp += nfpl[j];
      detect(cur, ch, cw, fx, fy, nkp, roff);
      roff += nkp;
    }
  }
  if (roff > MAXT) roff = MAXT;
  k_topk_final<<<1, 1024, 0, stream>>>(resp_all, lafs_all, roff, out);
}
